// PointEncoder_68264210202830
// MI455X (gfx1250) — matrix operand order repaired and checked by reading, not compiled, not run on hardware
//
#include <hip/hip_runtime.h>

// CDNA5 / gfx1250: wave32, WMMA 16x16x32 f16 -> f32 accumulate.
typedef __attribute__((ext_vector_type(16))) _Float16 v16h;
typedef __attribute__((ext_vector_type(8)))  _Float16 v8h;
typedef __attribute__((ext_vector_type(8)))  float    v8f;

#define BG   8      // graphs
#define NP   4096   // points per graph
#define S1C  1024   // SA1 sample count
#define S2C  256    // SA2 sample count
#define KNN1 32
#define KNN2 32

// ---------------------------------------------------------------------------
// Farthest point sampling: one workgroup per graph, LDS-resident distances.
// ---------------------------------------------------------------------------
__global__ void fps_kernel(const float* __restrict__ pos, int n, int nsample,
                           int* __restrict__ idx)
{
  __shared__ float dist[NP];
  __shared__ float sred[256];
  __shared__ int   sredi[256];
  const int b = blockIdx.x;
  const int t = threadIdx.x;
  const int T = blockDim.x;
  const float* P = pos + (size_t)b * n * 3;
  int* I = idx + (size_t)b * nsample;
  const float fx = P[0], fy = P[1], fz = P[2];
  for (int i = t; i < n; i += T) {
    float dx = P[i*3+0]-fx, dy = P[i*3+1]-fy, dz = P[i*3+2]-fz;
    dist[i] = dx*dx + dy*dy + dz*dz;
  }
  if (t == 0) I[0] = 0;
  __syncthreads();
  for (int s = 1; s < nsample; ++s) {
    float best = -1.0f; int bi = 0;
    for (int i = t; i < n; i += T)
      if (dist[i] > best) { best = dist[i]; bi = i; }
    sred[t] = best; sredi[t] = bi;
    __syncthreads();
    for (int off = T >> 1; off > 0; off >>= 1) {
      if (t < off && sred[t+off] > sred[t]) { sred[t] = sred[t+off]; sredi[t] = sredi[t+off]; }
      __syncthreads();
    }
    const int nxt = sredi[0];
    if (t == 0) I[s] = nxt;
    const float nx = P[nxt*3+0], ny = P[nxt*3+1], nz = P[nxt*3+2];
    __syncthreads();
    for (int i = t; i < n; i += T) {
      float dx = P[i*3+0]-nx, dy = P[i*3+1]-ny, dz = P[i*3+2]-nz;
      float d = dx*dx + dy*dy + dz*dz;
      if (d < dist[i]) dist[i] = d;
    }
    __syncthreads();
  }
}

// ---------------------------------------------------------------------------
// kNN: thread-per-query, register-resident insertion-sorted top-K.
// ---------------------------------------------------------------------------
template<int KN>
__global__ void knn_kernel(const float* __restrict__ Q, const float* __restrict__ S,
                           int nq, int ns, int* __restrict__ oidx, float* __restrict__ od2)
{
  const int q = blockIdx.x * blockDim.x + threadIdx.x;
  if (q >= BG * nq) return;
  const int b = q / nq;
  const float qx = Q[(size_t)q*3+0], qy = Q[(size_t)q*3+1], qz = Q[(size_t)q*3+2];
  const float* Sp = S + (size_t)b * ns * 3;
  float bd[KN]; int bi[KN];
#pragma unroll
  for (int t = 0; t < KN; ++t) { bd[t] = 3.4e38f; bi[t] = 0; }
  for (int i = 0; i < ns; ++i) {
    const float dx = Sp[i*3+0]-qx, dy = Sp[i*3+1]-qy, dz = Sp[i*3+2]-qz;
    float cd = dx*dx + dy*dy + dz*dz;
    int ci = i;
#pragma unroll
    for (int t = 0; t < KN; ++t) {
      if (cd < bd[t]) { float td = bd[t]; int ti = bi[t]; bd[t] = cd; bi[t] = ci; cd = td; ci = ti; }
    }
  }
#pragma unroll
  for (int t = 0; t < KN; ++t) {
    oidx[(size_t)q*KN + t] = bi[t];
    if (od2) od2[(size_t)q*KN + t] = fmaxf(bd[t], 0.0f);
  }
}

// ---------------------------------------------------------------------------
// Row gather (f32): out[b,s,:] = in[b, idx[b,s], :]
// ---------------------------------------------------------------------------
__global__ void gather_rows(const float* __restrict__ in, const int* __restrict__ idx,
                            float* __restrict__ out, int ns, int n, int C)
{
  const long long t = (long long)blockIdx.x * blockDim.x + threadIdx.x;
  if (t >= (long long)BG * ns * C) return;
  const int c = (int)(t % C);
  const int s = (int)((t / C) % ns);
  const int b = (int)(t / ((long long)C * ns));
  out[t] = in[((size_t)b * n + idx[b*ns + s]) * C + c];
}

// ---------------------------------------------------------------------------
// PointNetConv message features -> f16 rows of width Kpad (tail zeroed):
// [x_j (Cx), pos_j - pos_i (3), 0...]
// ---------------------------------------------------------------------------
__global__ void build_feat(const float* __restrict__ xf, const float* __restrict__ pos,
                           const float* __restrict__ posd, const int* __restrict__ nbr,
                           _Float16* __restrict__ out, int S, int Kn, int n, int Cx, int Kpad)
{
  const int r = blockIdx.x * blockDim.x + threadIdx.x;
  if (r >= BG * S * Kn) return;
  const int b = r / (S * Kn);
  const int s = (r / Kn) % S;
  const int nb = nbr[r];
  const float* xs = xf   + ((size_t)b * n + nb) * Cx;
  const float* ps = pos  + ((size_t)b * n + nb) * 3;
  const float* pc = posd + ((size_t)b * S + s) * 3;
  _Float16* o = out + (size_t)r * Kpad;
  for (int c = 0; c < Cx; ++c) o[c] = (_Float16)xs[c];
  o[Cx+0] = (_Float16)(ps[0] - pc[0]);
  o[Cx+1] = (_Float16)(ps[1] - pc[1]);
  o[Cx+2] = (_Float16)(ps[2] - pc[2]);
  for (int c = Cx + 3; c < Kpad; ++c) o[c] = (_Float16)0.0f;
}

// ---------------------------------------------------------------------------
// Row-wise concat -> f16 width Kpad (tail zeroed); A rows broadcast by agroup.
// ---------------------------------------------------------------------------
__global__ void concat2_f16(const float* __restrict__ A, int Ca, int agroup,
                            const float* __restrict__ Bv, int Cb,
                            _Float16* __restrict__ out, int rows, int Kpad)
{
  const int r = blockIdx.x * blockDim.x + threadIdx.x;
  if (r >= rows) return;
  const float* ar = A + (size_t)(agroup > 1 ? r / agroup : r) * Ca;
  _Float16* o = out + (size_t)r * Kpad;
  for (int c = 0; c < Ca; ++c) o[c] = (_Float16)ar[c];
  const float* br = Bv + (size_t)r * Cb;
  for (int c = 0; c < Cb; ++c) o[Ca + c] = (_Float16)br[c];
  for (int c = Ca + Cb; c < Kpad; ++c) o[c] = (_Float16)0.0f;
}

// ---------------------------------------------------------------------------
// f32 -> padded f16 rows (for the skip-connection GEMM inputs).
// ---------------------------------------------------------------------------
__global__ void cvt_pad_f16(const float* __restrict__ in, _Float16* __restrict__ out,
                            long long rows, int Cin, int Kpad)
{
  const long long i = (long long)blockIdx.x * blockDim.x + threadIdx.x;
  if (i >= rows * Kpad) return;
  const int c = (int)(i % Kpad);
  const long long r = i / Kpad;
  out[i] = (c < Cin) ? (_Float16)in[r * Cin + c] : (_Float16)0.0f;
}

// ---------------------------------------------------------------------------
// Pack f32 weights [K,N] into f16 WMMA-B lane layout, K padded to Kpad:
// Wp[((kc/32)*N + n)*32 + (k - kc)] so each column's 32 halves per k-chunk are contiguous.
// ---------------------------------------------------------------------------
__global__ void pack_weights(const float* __restrict__ W, _Float16* __restrict__ Wp,
                             int K, int Nc, long long total)
{
  const long long i = (long long)blockIdx.x * blockDim.x + threadIdx.x;
  if (i >= total) return;
  const int off32 = (int)(i & 31);
  const long long grp = i >> 5;
  const int n  = (int)(grp % Nc);
  const int kc = (int)(grp / Nc) << 5;
  const int k  = kc + off32;
  Wp[i] = (k < K) ? (_Float16)W[(size_t)k * Nc + n] : (_Float16)0.0f;
}

__global__ void fill_u32(unsigned int* __restrict__ p, unsigned int v, long long n)
{
  const long long i = (long long)blockIdx.x * blockDim.x + threadIdx.x;
  if (i < n) p[i] = v;
}

__global__ void add_inplace(float* __restrict__ a, const float* __restrict__ b, long long n)
{
  const long long i = (long long)blockIdx.x * blockDim.x + threadIdx.x;
  if (i < n) a[i] += b[i];
}

// ---------------------------------------------------------------------------
// knn_interpolate: inverse-square-distance weighted average of 3 coarse rows.
// ---------------------------------------------------------------------------
__global__ void knn_interp(const float* __restrict__ src, const int* __restrict__ idx,
                           const float* __restrict__ d2, float* __restrict__ out,
                           int nqPerB, int srcPerB, int C)
{
  const int q = blockIdx.x * blockDim.x + threadIdx.x;
  if (q >= BG * nqPerB) return;
  const int b = q / nqPerB;
  float w0 = 1.0f / fmaxf(d2[(size_t)q*3+0], 1e-16f);
  float w1 = 1.0f / fmaxf(d2[(size_t)q*3+1], 1e-16f);
  float w2 = 1.0f / fmaxf(d2[(size_t)q*3+2], 1e-16f);
  const float inv = 1.0f / (w0 + w1 + w2);
  w0 *= inv; w1 *= inv; w2 *= inv;
  const float* r0 = src + ((size_t)b*srcPerB + idx[(size_t)q*3+0]) * C;
  const float* r1 = src + ((size_t)b*srcPerB + idx[(size_t)q*3+1]) * C;
  const float* r2 = src + ((size_t)b*srcPerB + idx[(size_t)q*3+2]) * C;
  float* o = out + (size_t)q * C;
  for (int c = 0; c < C; ++c) o[c] = w0*r0[c] + w1*r1[c] + w2*r2[c];
}

// f32 atomic max via ordered integer-bit trick (lowers to global_atomic_max/min).
__device__ __forceinline__ void atomic_max_f32(float* addr, float v)
{
  if (v >= 0.0f) atomicMax((int*)addr, __float_as_int(v));
  else           atomicMin((unsigned int*)addr, __float_as_uint(v));
}

// ---------------------------------------------------------------------------
// Wave-per-tile WMMA GEMM: Y = act(A[M,Kpad] @ Wp + bias).
// A: f16 row-major, K padded to multiple of 32 (zero pad), M multiple of 16.
// Wp: f16 pre-packed B lane layout (see pack_weights).
// Inner loop: 4x global_load_b128 + 1x v_wmma. No guards, no conversions.
//   SHALF: store f16; AMAX: grouped atomic-max f32 (Y pre-filled with -inf).
// ---------------------------------------------------------------------------
template<bool SHALF, bool RELU, bool AMAX>
__global__ void gemm_wmma(const _Float16* __restrict__ Ah, const _Float16* __restrict__ Wp,
                          const float* __restrict__ bias, void* __restrict__ Yp,
                          int Kpad, int Nc, int tilesN, long long totalTiles, int group)
{
  const long long gw = (long long)blockIdx.x * (blockDim.x >> 5) + (threadIdx.x >> 5);
  if (gw >= totalTiles) return;               // wave-uniform exit (EXEC all-ones at WMMA)
  const int lane = threadIdx.x & 31;
  const int half = lane >> 4;
  const int l    = lane & 15;
  const int tm   = (int)(gw / tilesN) * 16;
  const int tn   = (int)(gw % tilesN) * 16;
  // A lane layout (16x32 f16): row = tm + l; K = {kc+half*8 ..+8} U {kc+16+half*8 ..+8}
  const _Float16* ap = Ah + (size_t)(tm + l) * Kpad + half * 8;
  // B lane layout: col = tn + l; K = {kc+half*8 ..+8} U {kc+16+half*8 ..+8}
  const _Float16* wp = Wp + ((size_t)(tn + l)) * 32 + half * 8;
  const size_t wstride = (size_t)Nc * 32;
  v8f acc = {};
  for (int kc = 0; kc < Kpad; kc += 32) {
    const v8h a0 = *(const v8h*)(ap);
    const v8h a1 = *(const v8h*)(ap + 16);
    const v8h b0 = *(const v8h*)(wp);
    const v8h b1 = *(const v8h*)(wp + 16);
    v16h a, b;
#pragma unroll
    for (int t = 0; t < 8; ++t) {
      a[t] = a0[t]; a[t + 8] = a1[t];
      b[t] = b0[t]; b[t + 8] = b1[t];
    }
    acc = __builtin_amdgcn_wmma_f32_16x16x32_f16(false, a, false, b, (short)0, acc,
                                                 false, false);
    ap += 32;
    wp += wstride;
  }
  const float bb = bias[tn + l];
#pragma unroll
  for (int r = 0; r < 8; ++r) {
    const int row = tm + r + half * 8;          // D: (M = r + 8*(lane/16), N = lane%16)
    float v = acc[r] + bb;
    if (RELU) v = fmaxf(v, 0.0f);
    if (AMAX) {
      atomic_max_f32((float*)Yp + (size_t)(row / group) * Nc + (tn + l), v);
    } else if (SHALF) {
      ((_Float16*)Yp)[(size_t)row * Nc + (tn + l)] = (_Float16)v;
    } else {
      ((float*)Yp)[(size_t)row * Nc + (tn + l)] = v;
    }
  }
}

// ---------------------------------------------------------------------------
extern "C" void kernel_launch(void* const* d_in, const int* in_sizes, int n_in,
                              void* d_out, int out_size, void* d_ws, size_t ws_size,
                              hipStream_t stream)
{
  (void)in_sizes; (void)n_in; (void)out_size; (void)ws_size;
  // Inputs flattened in setup_inputs() insertion order:
  // 0:x 1:pos 2:batch, then params leaves (W,b per layer, module insertion order).
  const float* x   = (const float*)d_in[0];
  const float* pos = (const float*)d_in[1];
  const float* W_sa1n0 = (const float*)d_in[3],  * b_sa1n0 = (const float*)d_in[4];
  const float* W_sa1n1 = (const float*)d_in[5],  * b_sa1n1 = (const float*)d_in[6];
  const float* W_sa1l  = (const float*)d_in[7],  * b_sa1l  = (const float*)d_in[8];
  const float* W_sa2n0 = (const float*)d_in[9],  * b_sa2n0 = (const float*)d_in[10];
  const float* W_sa2n1 = (const float*)d_in[11], * b_sa2n1 = (const float*)d_in[12];
  const float* W_sa2l  = (const float*)d_in[13], * b_sa2l  = (const float*)d_in[14];
  const float* W_g0    = (const float*)d_in[15], * b_g0    = (const float*)d_in[16];
  const float* W_g1    = (const float*)d_in[17], * b_g1    = (const float*)d_in[18];
  const float* W_fp10  = (const float*)d_in[19], * b_fp10  = (const float*)d_in[20];
  const float* W_fp11  = (const float*)d_in[21], * b_fp11  = (const float*)d_in[22];
  const float* W_fp20  = (const float*)d_in[23], * b_fp20  = (const float*)d_in[24];
  const float* W_fp21  = (const float*)d_in[25], * b_fp21  = (const float*)d_in[26];
  const float* W_fp30  = (const float*)d_in[27], * b_fp30  = (const float*)d_in[28];
  const float* W_fp31  = (const float*)d_in[29], * b_fp31  = (const float*)d_in[30];
  const float* W_fp32  = (const float*)d_in[31], * b_fp32  = (const float*)d_in[32];

  // ---- workspace bump allocator (persistent block + reusable arena) ----
  char* base = (char*)d_ws;
  size_t off = 0;
  auto alloc = [&](size_t bytes) -> void* {
    void* p = base + off;
    off += (bytes + 255) & ~(size_t)255;
    return p;
  };
  auto nblk = [](long long n, int t) { return dim3((unsigned)((n + t - 1) / t)); };

  // Persistent buffers
  int*   idx1 = (int*)  alloc((size_t)BG*S1C*4);
  float* p1   = (float*)alloc((size_t)BG*S1C*3*4);
  float* xd1  = (float*)alloc((size_t)BG*S1C*3*4);
  float* x1   = (float*)alloc((size_t)BG*S1C*128*4);
  int*   nbr1 = (int*)  alloc((size_t)BG*S1C*KNN1*4);
  int*   idx2 = (int*)  alloc((size_t)BG*S2C*4);
  float* p2   = (float*)alloc((size_t)BG*S2C*3*4);
  float* x2   = (float*)alloc((size_t)BG*S2C*256*4);
  int*   nbr2 = (int*)  alloc((size_t)BG*S2C*KNN2*4);
  float* gbuf = (float*)alloc((size_t)BG*512*4);
  float* hA   = (float*)alloc((size_t)BG*S2C*256*4);
  float* hB   = (float*)alloc((size_t)BG*S1C*128*4);
  int*   idxA = (int*)  alloc((size_t)BG*S1C*3*4);
  float* d2A  = (float*)alloc((size_t)BG*S1C*3*4);
  int*   idxB = (int*)  alloc((size_t)BG*NP*3*4);
  float* d2B  = (float*)alloc((size_t)BG*NP*3*4);

  // Pre-packed f16 weights (WMMA B lane layout), persistent (~2 MB total).
  auto pack = [&](const float* W, int K, int Nc, int Kpad) -> const _Float16* {
    _Float16* Wp = (_Float16*)alloc((size_t)Kpad * Nc * 2);
    const long long total = (long long)Kpad * Nc;
    pack_weights<<<nblk(total, 256), 256, 0, stream>>>(W, Wp, K, Nc, total);
    return Wp;
  };
  const _Float16* P_sa1n0 = pack(W_sa1n0,   6,  64,  32);
  const _Float16* P_sa1n1 = pack(W_sa1n1,  64, 128,  64);
  const _Float16* P_sa1l  = pack(W_sa1l,    3, 128,  32);
  const _Float16* P_sa2n0 = pack(W_sa2n0, 131, 128, 160);
  const _Float16* P_sa2n1 = pack(W_sa2n1, 128, 256, 128);
  const _Float16* P_sa2l  = pack(W_sa2l,  128, 256, 128);
  const _Float16* P_g0    = pack(W_g0,    259, 256, 288);
  const _Float16* P_g1    = pack(W_g1,    256, 512, 256);
  const _Float16* P_fp10  = pack(W_fp10,  768, 512, 768);
  const _Float16* P_fp11  = pack(W_fp11,  512, 256, 512);
  const _Float16* P_fp20  = pack(W_fp20,  384, 256, 384);
  const _Float16* P_fp21  = pack(W_fp21,  256, 128, 256);
  const _Float16* P_fp30  = pack(W_fp30,  131, 128, 160);
  const _Float16* P_fp31  = pack(W_fp31,  128, 128, 128);
  const _Float16* P_fp32  = pack(W_fp32,  128, 128, 128);

  const size_t arenaBase = off;

  // mode 0: f16 out + ReLU; mode 1: grouped atomic-max f32; mode 2: f32 out.
  auto gemm = [&](const _Float16* A, const _Float16* Wp, const float* bias, void* Y,
                  int M, int Kpad, int Nc, int mode, int group) {
    const int tilesN = Nc / 16;
    const long long totalTiles = (long long)(M / 16) * tilesN;  // M always multiple of 16
    dim3 g((unsigned)((totalTiles + 3) / 4)), blk(128);
    switch (mode) {
      case 0: gemm_wmma<true,  true,  false><<<g, blk, 0, stream>>>(A, Wp, bias, Y, Kpad, Nc, tilesN, totalTiles, group); break;
      case 1: gemm_wmma<false, false, true ><<<g, blk, 0, stream>>>(A, Wp, bias, Y, Kpad, Nc, tilesN, totalTiles, group); break;
      case 2: gemm_wmma<false, false, false><<<g, blk, 0, stream>>>(A, Wp, bias, Y, Kpad, Nc, tilesN, totalTiles, group); break;
    }
  };

  // ========================= SA1 =========================
  off = arenaBase;
  _Float16* feat1 = (_Float16*)alloc((size_t)BG*S1C*KNN1*32*2);
  _Float16* h1    = (_Float16*)alloc((size_t)BG*S1C*KNN1*64*2);
  _Float16* xd1h  = (_Float16*)alloc((size_t)BG*S1C*32*2);
  float*    s1    = (float*)   alloc((size_t)BG*S1C*128*4);

  fps_kernel<<<BG, 256, 0, stream>>>(pos, NP, S1C, idx1);
  gather_rows<<<nblk((long long)BG*S1C*3, 256), 256, 0, stream>>>(pos, idx1, p1, S1C, NP, 3);
  gather_rows<<<nblk((long long)BG*S1C*3, 256), 256, 0, stream>>>(x,   idx1, xd1, S1C, NP, 3);
  knn_kernel<KNN1><<<nblk((long long)BG*S1C, 128), 128, 0, stream>>>(p1, pos, S1C, NP, nbr1, nullptr);
  build_feat<<<nblk((long long)BG*S1C*KNN1, 256), 256, 0, stream>>>(x, pos, p1, nbr1, feat1, S1C, KNN1, NP, 3, 32);
  gemm(feat1, P_sa1n0, b_sa1n0, h1, BG*S1C*KNN1, 32, 64, /*mode=*/0, 1);             // 6->64 ReLU
  fill_u32<<<nblk((long long)BG*S1C*128, 256), 256, 0, stream>>>((unsigned*)x1, 0xFF800000u, (long long)BG*S1C*128);
  gemm(h1, P_sa1n1, b_sa1n1, x1, BG*S1C*KNN1, 64, 128, /*mode=*/1, KNN1);            // 64->128, max over k
  cvt_pad_f16<<<nblk((long long)BG*S1C*32, 256), 256, 0, stream>>>(xd1, xd1h, BG*S1C, 3, 32);
  gemm(xd1h, P_sa1l, b_sa1l, s1, BG*S1C, 32, 128, /*mode=*/2, 1);                    // skip 3->128
  add_inplace<<<nblk((long long)BG*S1C*128, 256), 256, 0, stream>>>(x1, s1, (long long)BG*S1C*128);

  // ========================= SA2 =========================
  off = arenaBase;
  _Float16* feat2 = (_Float16*)alloc((size_t)BG*S2C*KNN2*160*2);
  _Float16* h2    = (_Float16*)alloc((size_t)BG*S2C*KNN2*128*2);
  float*    xd2   = (float*)   alloc((size_t)BG*S2C*128*4);
  _Float16* xd2h  = (_Float16*)alloc((size_t)BG*S2C*128*2);
  float*    s2    = (float*)   alloc((size_t)BG*S2C*256*4);

  fps_kernel<<<BG, 256, 0, stream>>>(p1, S1C, S2C, idx2);
  gather_rows<<<nblk((long long)BG*S2C*3, 256), 256, 0, stream>>>(p1, idx2, p2, S2C, S1C, 3);
  gather_rows<<<nblk((long long)BG*S2C*128, 256), 256, 0, stream>>>(x1, idx2, xd2, S2C, S1C, 128);
  knn_kernel<KNN2><<<nblk((long long)BG*S2C, 128), 128, 0, stream>>>(p2, p1, S2C, S1C, nbr2, nullptr);
  build_feat<<<nblk((long long)BG*S2C*KNN2, 256), 256, 0, stream>>>(x1, p1, p2, nbr2, feat2, S2C, KNN2, S1C, 128, 160);
  gemm(feat2, P_sa2n0, b_sa2n0, h2, BG*S2C*KNN2, 160, 128, /*mode=*/0, 1);           // 131->128 ReLU
  fill_u32<<<nblk((long long)BG*S2C*256, 256), 256, 0, stream>>>((unsigned*)x2, 0xFF800000u, (long long)BG*S2C*256);
  gemm(h2, P_sa2n1, b_sa2n1, x2, BG*S2C*KNN2, 128, 256, /*mode=*/1, KNN2);           // 128->256, max over k
  cvt_pad_f16<<<nblk((long long)BG*S2C*128, 256), 256, 0, stream>>>(xd2, xd2h, BG*S2C, 128, 128);
  gemm(xd2h, P_sa2l, b_sa2l, s2, BG*S2C, 128, 256, /*mode=*/2, 1);                   // skip 128->256
  add_inplace<<<nblk((long long)BG*S2C*256, 256), 256, 0, stream>>>(x2, s2, (long long)BG*S2C*256);

  // ================= Global SA + FP1 =====================
  off = arenaBase;
  _Float16* gfeat   = (_Float16*)alloc((size_t)BG*S2C*288*2);
  _Float16* gh      = (_Float16*)alloc((size_t)BG*S2C*256*2);
  _Float16* fp1feat = (_Float16*)alloc((size_t)BG*S2C*768*2);
  _Float16* fp1h    = (_Float16*)alloc((size_t)BG*S2C*512*2);

  concat2_f16<<<nblk(BG*S2C, 128), 128, 0, stream>>>(x2, 256, 1, p2, 3, gfeat, BG*S2C, 288);
  gemm(gfeat, P_g0, b_g0, gh, BG*S2C, 288, 256, /*mode=*/0, 1);                      // 259->256 ReLU
  fill_u32<<<nblk((long long)BG*512, 256), 256, 0, stream>>>((unsigned*)gbuf, 0xFF800000u, (long long)BG*512);
  gemm(gh, P_g1, b_g1, gbuf, BG*S2C, 256, 512, /*mode=*/1, S2C);                     // 256->512, global max pool
  // FP1: k=1 interpolation from a single zero-pos coarse point == broadcast g[b]
  concat2_f16<<<nblk(BG*S2C, 128), 128, 0, stream>>>(gbuf, 512, S2C, x2, 256, fp1feat, BG*S2C, 768);
  gemm(fp1feat, P_fp10, b_fp10, fp1h, BG*S2C, 768, 512, /*mode=*/0, 1);              // 768->512 ReLU
  gemm(fp1h, P_fp11, b_fp11, hA, BG*S2C, 512, 256, /*mode=*/2, 1);                   // 512->256

  // ======================== FP2 ==========================
  off = arenaBase;
  float*    interpA = (float*)   alloc((size_t)BG*S1C*256*4);
  _Float16* fp2feat = (_Float16*)alloc((size_t)BG*S1C*384*2);
  _Float16* fp2h    = (_Float16*)alloc((size_t)BG*S1C*256*2);

  knn_kernel<3><<<nblk((long long)BG*S1C, 128), 128, 0, stream>>>(p1, p2, S1C, S2C, idxA, d2A);
  knn_interp<<<nblk(BG*S1C, 128), 128, 0, stream>>>(hA, idxA, d2A, interpA, S1C, S2C, 256);
  concat2_f16<<<nblk(BG*S1C, 128), 128, 0, stream>>>(interpA, 256, 1, x1, 128, fp2feat, BG*S1C, 384);
  gemm(fp2feat, P_fp20, b_fp20, fp2h, BG*S1C, 384, 256, /*mode=*/0, 1);              // 384->256 ReLU
  gemm(fp2h, P_fp21, b_fp21, hB, BG*S1C, 256, 128, /*mode=*/2, 1);                   // 256->128

  // ======================== FP3 ==========================
  off = arenaBase;
  float*    interpB = (float*)   alloc((size_t)BG*NP*128*4);
  _Float16* fp3feat = (_Float16*)alloc((size_t)BG*NP*160*2);
  _Float16* fh1     = (_Float16*)alloc((size_t)BG*NP*128*2);
  _Float16* fh2     = (_Float16*)alloc((size_t)BG*NP*128*2);

  knn_kernel<3><<<nblk((long long)BG*NP, 128), 128, 0, stream>>>(pos, p1, NP, S1C, idxB, d2B);
  knn_interp<<<nblk(BG*NP, 128), 128, 0, stream>>>(hB, idxB, d2B, interpB, NP, S1C, 128);
  concat2_f16<<<nblk(BG*NP, 128), 128, 0, stream>>>(interpB, 128, 1, x, 3, fp3feat, BG*NP, 160);
  gemm(fp3feat, P_fp30, b_fp30, fh1, BG*NP, 160, 128, /*mode=*/0, 1);                // 131->128 ReLU
  gemm(fh1, P_fp31, b_fp31, fh2, BG*NP, 128, 128, /*mode=*/0, 1);                    // 128->128 ReLU
  gemm(fh2, P_fp32, b_fp32, d_out, BG*NP, 128, 128, /*mode=*/2, 1);                  // 128->128 -> d_out (f32)
}
